// Hedger_54443005444410
// MI455X (gfx1250) — matrix operand order repaired and checked by reading, not compiled, not run on hardware
//
#include <hip/hip_runtime.h>
#include <hip/hip_bf16.h>

// ---------------------------------------------------------------------------
// Deep-hedging forward pass for MI455X (gfx1250), bf16 WMMA + f32 accumulate.
// B=8192 samples, T=64 steps, A=8 assets, F=16 features, H=512 hidden.
// Each workgroup owns MB=64 batch rows across all T steps (holding recurrence
// stays in LDS). All three layer GEMMs run on v_wmma_f32_16x16x32_bf16.
// LDS row strides are padded (520 / 40 halves) so A/B-fragment ds_load_b128s
// are bank-conflict-minimal (256-dword strides would 16-way serialize).
// ---------------------------------------------------------------------------

typedef __attribute__((ext_vector_type(16))) __bf16 bf16x16;
typedef __attribute__((ext_vector_type(8)))  float  f32x8;

namespace {
constexpr int kB   = 8192;
constexpr int kT   = 64;
constexpr int kA   = 8;
constexpr int kF   = 16;
constexpr int kH   = 512;
constexpr int kK1  = 32;           // layer-1 K padded 24 -> 32 (one WMMA)
constexpr int kMB  = 64;           // batch rows per workgroup
constexpr int kTP1 = kT + 1;

constexpr int kHS  = 520;          // H1s/H2s/W3s row stride (halves): 260 dw,
                                   // 260 % 64 = 4 -> conflict-minimal b128
constexpr int kXS  = 40;           // Xs/W1s row stride (halves): 20 dw % 64

// workspace layout (bytes)
constexpr size_t kW1tOff = 0;                                // 512*32  bf16 = 32 KB
constexpr size_t kW2tOff = kW1tOff + (size_t)kH * kK1 * 2;   // 512*512 bf16 = 512 KB
constexpr size_t kW3tOff = kW2tOff + (size_t)kH * kH * 2;    // 16*512  bf16 = 16 KB
}

// float -> bf16 bits via the native conversion (v_cvt_pk_bf16_f32, RNE)
__device__ __forceinline__ unsigned short f2bf(float f) {
  __bf16 h = (__bf16)f;
  unsigned short u;
  __builtin_memcpy(&u, &h, 2);
  return u;
}

__device__ __forceinline__ f32x8 wmma_bf16(bf16x16 a, bf16x16 b, f32x8 c) {
  // (neg_a, A, neg_b, B, c_mod, C, reuse_a, reuse_b)
  return __builtin_amdgcn_wmma_f32_16x16x32_bf16(false, a, false, b,
                                                 (short)0, c, false, false);
}

// A-fragment (16x32 bf16, MxK): lane's row data = 16 consecutive K
// at k0 + (lane<16 ? 0 : 16).
__device__ __forceinline__ bf16x16 load_a(const unsigned short* p0) {
  bf16x16 a;
  __builtin_memcpy(&a, p0, 32);
  return a;
}

// B-fragment (32x16 bf16, KxN): lane holds col N=lane%16, 16 consecutive K at
// k0 + (lane<16 ? 0 : 16). Weights are pre-transposed N-major so this is one
// contiguous 32-byte load.
__device__ __forceinline__ bf16x16 load_b(const unsigned short* p) {
  bf16x16 b;
  __builtin_memcpy(&b, p, 32);
  return b;
}

// ---------------------------------------------------------------------------
// Prep: f32 weights -> bf16, transposed to N-major [n][k], with padding.
//   W1 (24,512) -> W1t [512][32]  (k>=24 zero)
//   W2 (512,512)-> W2t [512][512]
//   W3 (512,8)  -> W3t [16][512]  (n>=8 zero)
// ---------------------------------------------------------------------------
__global__ void prep_weights(const float* __restrict__ W1,
                             const float* __restrict__ W2,
                             const float* __restrict__ W3,
                             unsigned short* __restrict__ W1t,
                             unsigned short* __restrict__ W2t,
                             unsigned short* __restrict__ W3t) {
  const int stride = (int)gridDim.x * (int)blockDim.x;
  const int t0 = (int)blockIdx.x * (int)blockDim.x + (int)threadIdx.x;
  for (int i = t0; i < kH * kK1; i += stride) {
    int n = i >> 5, k = i & 31;
    W1t[i] = (k < kF + kA) ? f2bf(W1[k * kH + n]) : (unsigned short)0;
  }
  for (int i = t0; i < kH * kH; i += stride) {
    int n = i >> 9, k = i & 511;
    W2t[i] = f2bf(W2[k * kH + n]);
  }
  for (int i = t0; i < 16 * kH; i += stride) {
    int n = i >> 9, k = i & 511;
    W3t[i] = (n < kA) ? f2bf(W3[k * kA + n]) : (unsigned short)0;
  }
}

// ---------------------------------------------------------------------------
// Main recurrent kernel. 256 threads = 8 waves. LDS ~201 KB (padded strides).
// ---------------------------------------------------------------------------
__global__ __launch_bounds__(256, 1) void hedger_fwd(
    const float* __restrict__ prices,        // (B, T+1, A)
    const float* __restrict__ info,          // (B, T, F)
    const unsigned short* __restrict__ W1t,  // [512][32] dense
    const unsigned short* __restrict__ W2t,  // [512][512] dense
    const unsigned short* __restrict__ W3t,  // [16][512] dense
    const float* __restrict__ b1,
    const float* __restrict__ b2,
    const float* __restrict__ b3,
    float* __restrict__ out)                 // (B, T+1)
{
  __shared__ __align__(16) unsigned short Xs[kMB * kXS];   //  5 KB concat input
  __shared__ __align__(16) unsigned short H1s[kMB * kHS];  // 65 KB
  __shared__ __align__(16) unsigned short H2s[kMB * kHS];  // 65 KB
  __shared__ __align__(16) unsigned short W1s[kH * kXS];   // 40 KB
  __shared__ __align__(16) unsigned short W3s[16 * kHS];   // 16.25 KB
  __shared__ float b1s[kH];
  __shared__ float b2s[kH];
  __shared__ float b3s[16];
  __shared__ float holds[kMB * 16];
  __shared__ float wealths[kMB * kA];

  const int tid  = (int)threadIdx.x;
  const int wave = tid >> 5;
  const int lane = tid & 31;
  const int lrow = lane & 15;
  const int ahalf = (lane & 16) ? 16 : 0;   // A-frag K sub-offset (halves)
  const int bhalf = (lane & 16) ? 16 : 0;   // B-frag K sub-offset (halves)
  const int roff  = (lane & 16) ? 8 : 0;    // D-matrix row offset
  const int b0 = (int)blockIdx.x * kMB;

  const f32x8 z8 = {0.f, 0.f, 0.f, 0.f, 0.f, 0.f, 0.f, 0.f};

  // ---- one-time fills (re-stride while copying into padded LDS) ----
  {
    const uint4* s1 = (const uint4*)W1t;     // 2048 chunks of 8 halves
    for (int i = tid; i < (kH * kK1) / 8; i += 256) {
      int n = i >> 2, c = i & 3;             // row, 8-half chunk within row
      *(uint4*)&W1s[n * kXS + c * 8] = s1[i];
    }
    const uint4* s3 = (const uint4*)W3t;     // 1024 chunks of 8 halves
    for (int i = tid; i < (16 * kH) / 8; i += 256) {
      int n = i >> 6, c = i & 63;
      *(uint4*)&W3s[n * kHS + c * 8] = s3[i];
    }
  }
  for (int i = tid; i < kH; i += 256) { b1s[i] = b1[i]; b2s[i] = b2[i]; }
  if (tid < 16) b3s[tid] = (tid < kA) ? b3[tid] : 0.0f;
  for (int i = tid; i < kMB * kA; i += 256) wealths[i] = 0.0f;
  // X(t=0): cols [0,16)=info, [16,24)=holding(=0), [24,32)=pad(=0)
  for (int i = tid; i < kMB * kK1; i += 256) {
    int m = i >> 5, c = i & 31;
    float v = (c < kF) ? info[(size_t)(b0 + m) * (kT * kF) + c] : 0.0f;
    Xs[m * kXS + c] = f2bf(v);
  }
  for (int i = tid; i < kMB; i += 256) out[(size_t)(b0 + i) * kTP1] = 0.0f;
  __syncthreads();

  for (int t = 0; t < kT; ++t) {
    // ---------------- layer 1: H1 = relu(X @ W1 + b1), K = 32 -------------
    {
      // A-fragments are shared by all 4 N-tiles of this wave: load once.
      bf16x16 af0 = load_a(&Xs[( 0 + lrow) * kXS + ahalf]);
      bf16x16 af1 = load_a(&Xs[(16 + lrow) * kXS + ahalf]);
      bf16x16 af2 = load_a(&Xs[(32 + lrow) * kXS + ahalf]);
      bf16x16 af3 = load_a(&Xs[(48 + lrow) * kXS + ahalf]);
      for (int nt = 0; nt < 4; ++nt) {
        const int n = (wave << 2) + nt;       // 8 waves * 4 = 32 col tiles
        const int col = (n << 4) + lrow;
        const bf16x16 bfrag = load_b(&W1s[col * kXS + bhalf]);
        const float bias = b1s[col];
        f32x8 acc[4];
        acc[0] = wmma_bf16(af0, bfrag, z8);
        acc[1] = wmma_bf16(af1, bfrag, z8);
        acc[2] = wmma_bf16(af2, bfrag, z8);
        acc[3] = wmma_bf16(af3, bfrag, z8);
        #pragma unroll
        for (int m = 0; m < 4; ++m) {
          const int rb = (m << 4) + roff;
          #pragma unroll
          for (int r = 0; r < 8; ++r)
            H1s[(rb + r) * kHS + col] = f2bf(fmaxf(acc[m][r] + bias, 0.0f));
        }
      }
    }
    __syncthreads();

    // ------- layer 2: H2 = relu(H1 @ W2 + b2), K = 512, 2 N-tiles/pass ----
    for (int half = 0; half < 2; ++half) {
      const int n0 = (wave << 2) + half * 2;  // tiles n0 and n0+1
      const int col0 = (n0 << 4) + lrow;
      const int col1 = col0 + 16;
      const unsigned short* wrow0 = W2t + (size_t)col0 * kH + bhalf;
      const unsigned short* wrow1 = W2t + (size_t)col1 * kH + bhalf;
      f32x8 acc00 = z8, acc01 = z8, acc02 = z8, acc03 = z8;  // N-tile n0
      f32x8 acc10 = z8, acc11 = z8, acc12 = z8, acc13 = z8;  // N-tile n0+1
      for (int k0 = 0; k0 < kH; k0 += 32) {
        const bf16x16 bf0 = load_b(wrow0 + k0);     // W2 read once per WG/step
        const bf16x16 bf1 = load_b(wrow1 + k0);
        const bf16x16 af0 = load_a(&H1s[( 0 + lrow) * kHS + k0 + ahalf]);
        const bf16x16 af1 = load_a(&H1s[(16 + lrow) * kHS + k0 + ahalf]);
        const bf16x16 af2 = load_a(&H1s[(32 + lrow) * kHS + k0 + ahalf]);
        const bf16x16 af3 = load_a(&H1s[(48 + lrow) * kHS + k0 + ahalf]);
        acc00 = wmma_bf16(af0, bf0, acc00);
        acc10 = wmma_bf16(af0, bf1, acc10);
        acc01 = wmma_bf16(af1, bf0, acc01);
        acc11 = wmma_bf16(af1, bf1, acc11);
        acc02 = wmma_bf16(af2, bf0, acc02);
        acc12 = wmma_bf16(af2, bf1, acc12);
        acc03 = wmma_bf16(af3, bf0, acc03);
        acc13 = wmma_bf16(af3, bf1, acc13);
      }
      const float bias0 = b2s[col0];
      const float bias1 = b2s[col1];
      f32x8 a0s[4] = {acc00, acc01, acc02, acc03};
      f32x8 a1s[4] = {acc10, acc11, acc12, acc13};
      #pragma unroll
      for (int m = 0; m < 4; ++m) {
        const int rb = (m << 4) + roff;
        #pragma unroll
        for (int r = 0; r < 8; ++r) {
          H2s[(rb + r) * kHS + col0] = f2bf(fmaxf(a0s[m][r] + bias0, 0.0f));
          H2s[(rb + r) * kHS + col1] = f2bf(fmaxf(a1s[m][r] + bias1, 0.0f));
        }
      }
    }
    __syncthreads();

    // ---------------- layer 3: holding = H2 @ W3 + b3 (N padded to 16) ----
    if (wave < 4) {
      const int m = wave;                     // 4 row tiles, 1 col tile
      f32x8 acc = z8;
      for (int k0 = 0; k0 < kH; k0 += 32) {
        const bf16x16 bfrag = load_b(&W3s[lrow * kHS + k0 + bhalf]);
        const bf16x16 afrag = load_a(&H2s[((m << 4) + lrow) * kHS + k0 + ahalf]);
        acc = wmma_bf16(afrag, bfrag, acc);
      }
      const float bias = b3s[lrow];
      const int rb = (m << 4) + roff;
      #pragma unroll
      for (int r = 0; r < 8; ++r)
        holds[(rb + r) * 16 + lrow] = acc[r] + bias;
    }
    __syncthreads();

    // ---------------- wealth update + assemble next-step input ------------
    for (int i = tid; i < kMB * kA; i += 256) {
      const int m = i >> 3, a = i & 7;
      const size_t pb = ((size_t)(b0 + m) * kTP1 + t) * kA + a;
      const float dp = prices[pb + kA] - prices[pb];
      const float h = holds[m * 16 + a];
      const float w = wealths[i] + h * dp;
      wealths[i] = w;
      Xs[m * kXS + kF + a] = f2bf(h);         // holding part of next input
    }
    if (t + 1 < kT) {
      for (int i = tid; i < kMB * kF; i += 256) {
        const int m = i >> 4, f = i & 15;
        Xs[m * kXS + f] =
            f2bf(info[((size_t)(b0 + m) * kT + (t + 1)) * kF + f]);
      }
    }
    __syncthreads();

    // out[b][t+1] = sum_a wealth  (reads only; next wealth write is 3
    // barriers away, next Xs read conflicts are also barrier-separated)
    for (int i = tid; i < kMB; i += 256) {
      float s = 0.0f;
      #pragma unroll
      for (int a = 0; a < kA; ++a) s += wealths[i * kA + a];
      out[(size_t)(b0 + i) * kTP1 + t + 1] = s;
    }
  }
}

// ---------------------------------------------------------------------------
extern "C" void kernel_launch(void* const* d_in, const int* in_sizes, int n_in,
                              void* d_out, int out_size, void* d_ws,
                              size_t ws_size, hipStream_t stream) {
  (void)in_sizes; (void)n_in; (void)out_size; (void)ws_size;
  const float* prices = (const float*)d_in[0];
  const float* info   = (const float*)d_in[1];
  // d_in[2] = payoff (unused in forward)
  const float* W1 = (const float*)d_in[3];
  const float* b1 = (const float*)d_in[4];
  const float* W2 = (const float*)d_in[5];
  const float* b2 = (const float*)d_in[6];
  const float* W3 = (const float*)d_in[7];
  const float* b3 = (const float*)d_in[8];
  float* out = (float*)d_out;

  char* ws = (char*)d_ws;
  unsigned short* W1t = (unsigned short*)(ws + kW1tOff);
  unsigned short* W2t = (unsigned short*)(ws + kW2tOff);
  unsigned short* W3t = (unsigned short*)(ws + kW3tOff);

  prep_weights<<<512, 256, 0, stream>>>(W1, W2, W3, W1t, W2t, W3t);
  hedger_fwd<<<kB / kMB, 256, 0, stream>>>(prices, info, W1t, W2t, W3t,
                                           b1, b2, b3, out);
}
